// SimpleMoE_71485435675244
// MI455X (gfx1250) — matrix operand order repaired and checked by reading, not compiled, not run on hardware
//
#include <hip/hip_runtime.h>

#define DIM  1024
#define HID  4096
#define NE   4
#define TOK  8192

typedef __attribute__((ext_vector_type(16))) __bf16 v16bf;
typedef __attribute__((ext_vector_type(8)))  float  v8f;
typedef __attribute__((ext_vector_type(4)))  unsigned int u32x4;

union Frag { v16bf v; u32x4 q[2]; };

// ---------------- workspace layout ----------------
// ints: [0..3] counts, [4..7] offsets, [8..11] cursors, [12..15] pad
//       [16 .. 16+TOK)      expert id per token
//       [16+TOK .. 16+2TOK) permuted token ids (grouped by expert)
// bytes: W1 swizzled bf16, W2 swizzled bf16, x converted to bf16
#define EID_OFF_I   16
#define PERM_OFF_I  (16 + TOK)
#define META_BYTES  (((16 + 2 * TOK) * 4 + 511) & ~511)
#define W1S_OFF_B   ((size_t)META_BYTES)
#define W1S_ELEMS   ((size_t)NE * DIM * HID)
#define W2S_OFF_B   (W1S_OFF_B + W1S_ELEMS * 2)
#define XBF_OFF_B   (W2S_OFF_B + W1S_ELEMS * 2)

// W1 tiles per expert: kt in [0,32) x nt in [0,256); W2: kt in [0,128) x nt in [0,64)
#define W1_TILES_PER_E (32 * 256)
#define W2_TILES_PER_E (128 * 64)

// ---------------- init: zero counters ----------------
__global__ void k_init(int* __restrict__ wsi) {
  if (threadIdx.x < 16) wsi[threadIdx.x] = 0;
}

// ---------------- router: logits -> argmax (softmax is monotone) ----------------
__global__ void k_router(const float* __restrict__ x, const float* __restrict__ Wr,
                         const float* __restrict__ br, int* __restrict__ wsi) {
  int wave = threadIdx.x >> 5, lane = threadIdx.x & 31;
  int t = blockIdx.x * 8 + wave;
  if (t >= TOK) return;
  const float*  xr  = x + (size_t)t * DIM;
  const float4* wr4 = (const float4*)Wr;   // Wr[k][e] row-major, 4 experts contiguous
  float a0 = 0.f, a1 = 0.f, a2 = 0.f, a3 = 0.f;
  for (int k = lane; k < DIM; k += 32) {
    float  xv = xr[k];
    float4 w  = wr4[k];
    a0 += xv * w.x; a1 += xv * w.y; a2 += xv * w.z; a3 += xv * w.w;
  }
  for (int off = 16; off; off >>= 1) {
    a0 += __shfl_xor(a0, off);
    a1 += __shfl_xor(a1, off);
    a2 += __shfl_xor(a2, off);
    a3 += __shfl_xor(a3, off);
  }
  if (lane == 0) {
    a0 += br[0]; a1 += br[1]; a2 += br[2]; a3 += br[3];
    int best = 0; float bv = a0;                 // first-max semantics like argmax
    if (a1 > bv) { bv = a1; best = 1; }
    if (a2 > bv) { bv = a2; best = 2; }
    if (a3 > bv) { bv = a3; best = 3; }
    wsi[EID_OFF_I + t] = best;
    atomicAdd(&wsi[best], 1);
  }
}

// ---------------- exclusive scan of 4 counts ----------------
__global__ void k_scan(int* __restrict__ wsi) {
  if (threadIdx.x == 0) {
    int o = 0;
    for (int e = 0; e < NE; e++) { wsi[4 + e] = o; o += wsi[e]; }
  }
}

// ---------------- scatter token ids grouped by expert ----------------
__global__ void k_scatter(int* __restrict__ wsi) {
  int t = blockIdx.x * 256 + threadIdx.x;
  if (t >= TOK) return;
  int e   = wsi[EID_OFF_I + t];
  int pos = atomicAdd(&wsi[8 + e], 1);
  wsi[PERM_OFF_I + wsi[4 + e] + pos] = t;
}

// ---------------- x fp32 -> bf16 (one-time, coalesced) ----------------
__global__ void k_xcvt(const float* __restrict__ x, __bf16* __restrict__ xb) {
  size_t i = ((size_t)blockIdx.x * 256 + threadIdx.x) * 4;
  float4 v = *(const float4*)(x + i);
  union { __bf16 h[4]; unsigned long long u; } p;
  p.h[0] = (__bf16)v.x; p.h[1] = (__bf16)v.y;
  p.h[2] = (__bf16)v.z; p.h[3] = (__bf16)v.w;
  *(unsigned long long*)(xb + i) = p.u;
}

// ---------------- weight swizzle: fp32 -> bf16 in WMMA B-fragment order ----------
// B (32x16 bf16): lane L: N = L&15, half = L>>4; per-lane j = 0..15 -> K = half*16 + j.
// Tile stored as 512 bf16: element [lane*16 + j]. Fragment load = 2x b128 per lane.
__global__ void k_swz(const float* __restrict__ W1, const float* __restrict__ W2,
                      __bf16* __restrict__ W1s, __bf16* __restrict__ W2s) {
  int gw   = blockIdx.x * 8 + (threadIdx.x >> 5);
  int lane = threadIdx.x & 31;
  int n    = lane & 15, half = lane >> 4;
  const float* src; __bf16* dst; int ld, kt, nt;
  if (gw < NE * W1_TILES_PER_E) {
    int e = gw / W1_TILES_PER_E, r = gw % W1_TILES_PER_E;
    kt = r / 256; nt = r % 256;
    src = W1 + (size_t)e * DIM * HID; ld = HID;
    dst = W1s + (size_t)gw * 512;
  } else {
    int g2 = gw - NE * W1_TILES_PER_E;
    int e = g2 / W2_TILES_PER_E, r = g2 % W2_TILES_PER_E;
    kt = r / 64; nt = r % 64;
    src = W2 + (size_t)e * HID * DIM; ld = DIM;
    dst = W2s + (size_t)g2 * 512;
  }
  const float* s = src + (size_t)(kt * 32 + half * 16) * ld + nt * 16 + n;
  Frag f;
#pragma unroll
  for (int j = 0; j < 16; j++) f.v[j] = (__bf16)s[(size_t)j * ld];
  u32x4* d = (u32x4*)(dst + lane * 16);
  d[0] = f.q[0];
  d[1] = f.q[1];
}

// ---------------- fused expert kernel --------------------------------------------
// Block = 32 tokens of one expert (two 16-row M-tiles), 256 threads = 8 waves.
// Wave w owns output columns [128w, 128w+128).  Hidden processed in 128-wide chunks:
//   GEMM1: wave w computes h[:, kc+16w .. +16) over full K=1024 (2x32 WMMAs, B reused
//          across both M-tiles), ReLU -> LDS
//   GEMM2: every wave consumes the shared 32x128 h chunk vs its 8 output n-tiles
//          (4 k-steps x 8 tiles x 2 M-tiles), accumulating in v8f[16] (128 VGPRs).
// x rows are gathered into LDS with CDNA5 async DMA (GLOBAL_LOAD_ASYNC_TO_LDS_B128,
// tracked by ASYNCcnt).
__global__ void __launch_bounds__(256)
k_moe(const __bf16* __restrict__ xb,
      const float* __restrict__ b1, const float* __restrict__ b2,
      const __bf16* __restrict__ W1s, const __bf16* __restrict__ W2s,
      const int* __restrict__ wsi, float* __restrict__ out) {
  int e    = blockIdx.y;
  int cnt  = wsi[e];
  int base = blockIdx.x * 32;
  if (base >= cnt) return;
  int off    = wsi[4 + e];
  int nvalid = cnt - base; if (nvalid > 32) nvalid = 32;

  __shared__ __bf16 xs[32 * DIM];   // 64 KB: x tile, bf16, row-major [m][k]
  __shared__ __bf16 hs[32 * 128];   //  8 KB: shared hidden chunk, row-major [m][c]
  __shared__ int    tok[32];

  int tid = threadIdx.x;
  if (tid < 32) {
    int i = tid < nvalid ? tid : nvalid - 1;          // clamp padding rows
    tok[tid] = wsi[PERM_OFF_I + off + base + i];
  }
  __syncthreads();

  // Gather token rows (already bf16) into LDS via async DMA: 32 rows x 2048 B
  // = 4096 16-byte chunks; each thread issues 16 GLOBAL_LOAD_ASYNC_TO_LDS_B128.
  {
    unsigned ldsbase = (unsigned)(size_t)&xs[0];
#pragma unroll
    for (int it = 0; it < 16; it++) {
      int c = tid + it * 256;                    // chunk id
      int m = c >> 7, k16 = c & 127;             // 128 x 16B per row
      unsigned ldsoff = ldsbase + (unsigned)(m * 2048 + k16 * 16);
      unsigned long long gaddr =
          (unsigned long long)(size_t)(xb + (size_t)tok[m] * DIM) + (unsigned)(k16 * 16);
      asm volatile("global_load_async_to_lds_b128 %0, %1, off"
                   :: "v"(ldsoff), "v"(gaddr) : "memory");
    }
    asm volatile("s_wait_asynccnt 0" ::: "memory");
  }
  __syncthreads();

  int wave = tid >> 5, lane = tid & 31;
  int half = lane >> 4, nn = lane & 15;

  v8f accY[16];                                  // [mt][nt] accumulators
#pragma unroll
  for (int i = 0; i < 16; i++) accY[i] = v8f{};

  const __bf16* w1e = W1s + (size_t)e * W1_TILES_PER_E * 512;
  const __bf16* w2e = W2s + (size_t)e * W2_TILES_PER_E * 512;

  for (int kc = 0; kc < HID; kc += 128) {
    // ---------- GEMM1: h columns [kc+16*wave, +16) for both M-tiles ----------
    int hnt = (kc >> 4) + wave;          // n-tile index into W1s (0..255)
    v8f acc1a = v8f{}, acc1b = v8f{};
#pragma unroll 2
    for (int k0 = 0; k0 < DIM; k0 += 32) {
      Frag a0, a1, b;
      const __bf16* bp = w1e + ((size_t)((k0 >> 5) * 256 + hnt)) * 512 + lane * 16;
      b.q[0] = *(const u32x4*)bp;
      b.q[1] = *(const u32x4*)(bp + 8);
      // A 16x32 bf16: lane=M; j0..15 -> K = k0+half*16+j (matches B swizzle)
      const __bf16* ap0 = xs + (size_t)nn * DIM + k0 + half * 16;
      a0.q[0] = *(const u32x4*)ap0;
      a0.q[1] = *(const u32x4*)(ap0 + 8);
      const __bf16* ap1 = ap0 + 16 * DIM;
      a1.q[0] = *(const u32x4*)ap1;
      a1.q[1] = *(const u32x4*)(ap1 + 8);
      acc1a = __builtin_amdgcn_wmma_f32_16x16x32_bf16(false, a0.v, false, b.v,
                                                      (short)0, acc1a, false, false);
      acc1b = __builtin_amdgcn_wmma_f32_16x16x32_bf16(false, a1.v, false, b.v,
                                                      (short)0, acc1b, false, false);
    }
    // bias + ReLU, C layout: row = half*8 + r, col = nn (fixed per lane)
    float bias = b1[e * HID + kc + wave * 16 + nn];
#pragma unroll
    for (int r = 0; r < 8; r++) {
      float v0 = acc1a[r] + bias; v0 = v0 > 0.f ? v0 : 0.f;
      float v1 = acc1b[r] + bias; v1 = v1 > 0.f ? v1 : 0.f;
      hs[(size_t)(half * 8 + r) * 128 + wave * 16 + nn]        = (__bf16)v0;
      hs[(size_t)(16 + half * 8 + r) * 128 + wave * 16 + nn]   = (__bf16)v1;
    }
    __syncthreads();

    // ---------- GEMM2: accumulate this hidden chunk into y ----------
#pragma unroll
    for (int kk = 0; kk < 128; kk += 32) {
      Frag a0, a1;
      const __bf16* ap0 = hs + (size_t)nn * 128 + kk + half * 16;
      a0.q[0] = *(const u32x4*)ap0;
      a0.q[1] = *(const u32x4*)(ap0 + 8);
      const __bf16* ap1 = ap0 + 16 * 128;
      a1.q[0] = *(const u32x4*)ap1;
      a1.q[1] = *(const u32x4*)(ap1 + 8);
      int kt2 = (kc + kk) >> 5;
#pragma unroll
      for (int nt = 0; nt < 8; nt++) {
        Frag b;
        const __bf16* bp = w2e + ((size_t)(kt2 * 64 + wave * 8 + nt)) * 512 + lane * 16;
        b.q[0] = *(const u32x4*)bp;
        b.q[1] = *(const u32x4*)(bp + 8);
        accY[nt] = __builtin_amdgcn_wmma_f32_16x16x32_bf16(false, a0.v, false, b.v,
                                                           (short)0, accY[nt], false, false);
        accY[8 + nt] = __builtin_amdgcn_wmma_f32_16x16x32_bf16(false, a1.v, false, b.v,
                                                               (short)0, accY[8 + nt], false, false);
      }
    }
    __syncthreads();   // hs reused next chunk
  }

  // ---------- epilogue: + b2, scatter rows back ----------
#pragma unroll
  for (int nt = 0; nt < 8; nt++) {
    int col = wave * 128 + nt * 16 + nn;
    float bb = b2[e * DIM + col];
#pragma unroll
    for (int mt = 0; mt < 2; mt++) {
#pragma unroll
      for (int r = 0; r < 8; r++) {
        int m = mt * 16 + half * 8 + r;
        if (m < nvalid) out[(size_t)tok[m] * DIM + col] = accY[mt * 8 + nt][r] + bb;
      }
    }
  }
}

// ---------------- host-side launch ----------------
extern "C" void kernel_launch(void* const* d_in, const int* in_sizes, int n_in,
                              void* d_out, int out_size, void* d_ws, size_t ws_size,
                              hipStream_t stream) {
  const float* x  = (const float*)d_in[0];
  const float* Wr = (const float*)d_in[1];
  const float* br = (const float*)d_in[2];
  const float* W1 = (const float*)d_in[3];
  const float* b1 = (const float*)d_in[4];
  const float* W2 = (const float*)d_in[5];
  const float* b2 = (const float*)d_in[6];
  float* out = (float*)d_out;
  int*    wsi = (int*)d_ws;
  __bf16* W1s = (__bf16*)((char*)d_ws + W1S_OFF_B);
  __bf16* W2s = (__bf16*)((char*)d_ws + W2S_OFF_B);
  __bf16* xbf = (__bf16*)((char*)d_ws + XBF_OFF_B);

  k_init   <<<1, 64, 0, stream>>>(wsi);
  k_router <<<TOK / 8, 256, 0, stream>>>(x, Wr, br, wsi);
  k_scan   <<<1, 32, 0, stream>>>(wsi);
  k_scatter<<<TOK / 256, 256, 0, stream>>>(wsi);
  k_xcvt   <<<(TOK * DIM / 4) / 256, 256, 0, stream>>>(x, xbf);
  k_swz    <<<(NE * (W1_TILES_PER_E + W2_TILES_PER_E)) / 8, 256, 0, stream>>>(W1, W2, W1s, W2s);
  dim3 grid(TOK / 32, NE);
  k_moe    <<<grid, 256, 0, stream>>>(xbf, b1, b2, W1s, W2s, wsi, out);
}
